// MultiHeadedAttention_9019431321633
// MI455X (gfx1250) — matrix operand order repaired and checked by reading, not compiled, not run on hardware
//
#include <hip/hip_runtime.h>
#include <hip/hip_bf16.h>
#include <stdint.h>

// ---------------------------------------------------------------------------
// MI455X (gfx1250) implementation. Memory-bound workload (~80 GFLOP vs ~0.5GB
// traffic @ 23.3 TB/s => ~20us floor), so: bf16 intermediates (halve HBM
// bytes), all GEMM-shaped math on v_wmma_f32_16x16x32_bf16 (f32 accumulate),
// softmax/reductions in fp32 VALU. wave32 everywhere.
// ---------------------------------------------------------------------------

typedef __attribute__((ext_vector_type(16))) __bf16 v16bf;
typedef __attribute__((ext_vector_type(8)))  float  v8f;

union Frag16 {
    uint4 q[2];
    v16bf v;
};

__device__ __forceinline__ unsigned short f2bf(float f) {
    union { float f; unsigned u; } x; x.f = f;
    unsigned u = x.u;
    u += 0x7fffu + ((u >> 16) & 1u);          // round-to-nearest-even
    return (unsigned short)(u >> 16);
}
__device__ __forceinline__ float bf2f(unsigned short h) {
    union { unsigned u; float f; } x; x.u = ((unsigned)h) << 16;
    return x.f;
}
__device__ __forceinline__ unsigned pack2(float a, float b) {
    return (unsigned)f2bf(a) | ((unsigned)f2bf(b) << 16);
}
__device__ __forceinline__ v8f v8f_zero() {
    v8f z = {0.f,0.f,0.f,0.f,0.f,0.f,0.f,0.f};
    return z;
}
__device__ __forceinline__ v8f wmma_bf16(const Frag16& a, const Frag16& b, v8f c) {
    return __builtin_amdgcn_wmma_f32_16x16x32_bf16(false, a.v, false, b.v,
                                                   (short)0, c, false, false);
}

// ---------------------------------------------------------------------------
// Generic GEMM: C[M,N] = Act[M,K] * W[N,K]^T + bias[N]
//   Act rows may be a concat of padRows pad-rows + main rows per batch.
//   Block tile 128x128, BK=32. 256 threads = 8 waves, wave tile 64x32.
// ---------------------------------------------------------------------------
template<bool ACT_BF16, bool OUT_F32>
__global__ __launch_bounds__(256) void gemm_wmma_kernel(
    const void* __restrict__ actMain, const float* __restrict__ actPad,
    int padRows, int rowsPerBatch,
    const float* __restrict__ W, const float* __restrict__ bias,
    void* __restrict__ outP, int M, int N, int K)
{
    __shared__ __align__(16) unsigned short Asl[128][40];   // +8 pad, 80B row
    __shared__ __align__(16) unsigned short Bsl[128][40];

    const int tid  = threadIdx.x;
    const int lane = tid & 31;
    const int wave = tid >> 5;
    const int mBlock = blockIdx.x * 128;
    const int nBlock = blockIdx.y * 128;

    // Per-thread loader rows: 4 (row,c4) slots each for A and B tiles.
    const float*          aRowF[4];
    const unsigned short* aRowH[4];
    const float*          bRowF[4];
    #pragma unroll
    for (int i = 0; i < 4; ++i) {
        int idx = tid + i * 256;
        int row = idx >> 3;            // 0..127 within tile
        int rg  = mBlock + row;        // global act row
        aRowF[i] = nullptr; aRowH[i] = nullptr;
        if (ACT_BF16) {
            aRowH[i] = (const unsigned short*)actMain + (size_t)rg * K;
        } else if (padRows > 0) {
            int nb = rg / rowsPerBatch;
            int s  = rg - nb * rowsPerBatch;
            if (s < padRows)
                aRowF[i] = actPad + ((size_t)nb * padRows + s) * K;
            else
                aRowF[i] = (const float*)actMain +
                           ((size_t)nb * (rowsPerBatch - padRows) + (s - padRows)) * K;
        } else {
            aRowF[i] = (const float*)actMain + (size_t)rg * K;
        }
        bRowF[i] = W + (size_t)(nBlock + row) * K;
    }

    const int wr = wave >> 2;          // 0..1 (M)
    const int wc = wave & 3;           // 0..3 (N)
    const int mW = wr * 64;
    const int nW = wc * 32;
    const int fm = lane & 15;
    const int lh = lane >> 4;
    const int kA = lh * 8;
    const int kB = lh * 8;

    v8f acc[4][2];
    #pragma unroll
    for (int i = 0; i < 4; ++i)
        #pragma unroll
        for (int j = 0; j < 2; ++j) acc[i][j] = v8f_zero();

    for (int kb = 0; kb < K; kb += 32) {
        __syncthreads();
        #pragma unroll
        for (int i = 0; i < 4; ++i) {
            int idx = tid + i * 256;
            int row = idx >> 3;
            int c4  = (idx & 7) * 4;
            uint2 pa;
            if (ACT_BF16) {
                pa = *(const uint2*)(aRowH[i] + kb + c4);
            } else {
                float4 f = *(const float4*)(aRowF[i] + kb + c4);
                pa.x = pack2(f.x, f.y); pa.y = pack2(f.z, f.w);
            }
            *(uint2*)&Asl[row][c4] = pa;
            float4 g = *(const float4*)(bRowF[i] + kb + c4);
            uint2 pb; pb.x = pack2(g.x, g.y); pb.y = pack2(g.z, g.w);
            *(uint2*)&Bsl[row][c4] = pb;
        }
        __syncthreads();

        Frag16 af[4], bf[2];
        #pragma unroll
        for (int i = 0; i < 4; ++i) {
            const unsigned short* p = &Asl[mW + i * 16 + fm][kA];
            af[i].q[0] = *(const uint4*)p;
            af[i].q[1] = *(const uint4*)(p + 16);
        }
        #pragma unroll
        for (int j = 0; j < 2; ++j) {
            const unsigned short* p = &Bsl[nW + j * 16 + fm][kB];
            bf[j].q[0] = *(const uint4*)p;
            bf[j].q[1] = *(const uint4*)(p + 16);
        }
        #pragma unroll
        for (int i = 0; i < 4; ++i)
            #pragma unroll
            for (int j = 0; j < 2; ++j)
                acc[i][j] = wmma_bf16(af[i], bf[j], acc[i][j]);
    }

    // Epilogue: C/D layout lane n = lane&15, VGPR r -> m = r + 8*(lane>>4)
    #pragma unroll
    for (int j = 0; j < 2; ++j) {
        int col = nBlock + nW + j * 16 + fm;
        float bval = bias[col];
        #pragma unroll
        for (int i = 0; i < 4; ++i) {
            int mrow = mBlock + mW + i * 16 + lh * 8;
            #pragma unroll
            for (int r = 0; r < 8; ++r) {
                float vv = acc[i][j][r] + bval;
                size_t off = (size_t)(mrow + r) * N + col;
                if (OUT_F32) ((float*)outP)[off] = vv;
                else ((unsigned short*)outP)[off] = f2bf(vv);
            }
        }
    }
}

// ---------------------------------------------------------------------------
// Local windowed attention: one wave per (n, s) pair.
//   y: bf16 (256, 132, 512) projected+padded; out: bf16 (256, 128, 512)
// ---------------------------------------------------------------------------
__global__ __launch_bounds__(256) void local_attn_kernel(
    const unsigned short* __restrict__ y, unsigned short* __restrict__ outq)
{
    const int lane = threadIdx.x & 31;
    const int wave = threadIdx.x >> 5;
    const int p = blockIdx.x * 8 + wave;   // 0..32767
    const int n = p >> 7;
    const int s = p & 127;
    const int d0 = lane * 16;
    const unsigned short* base = y + ((size_t)n * 132 + s) * 512 + d0;

    float w[5][16];
    #pragma unroll
    for (int j = 0; j < 5; ++j) {
        const unsigned short* pr = base + (size_t)j * 512;
        uint4 a = *(const uint4*)pr;
        uint4 b = *(const uint4*)(pr + 8);
        unsigned uu[8] = {a.x, a.y, a.z, a.w, b.x, b.y, b.z, b.w};
        #pragma unroll
        for (int t = 0; t < 8; ++t) {
            w[j][2*t]   = bf2f((unsigned short)(uu[t] & 0xffffu));
            w[j][2*t+1] = bf2f((unsigned short)(uu[t] >> 16));
        }
    }
    // window j=4 is the center row (padded index s+4 == original position s)
    float dots[5];
    #pragma unroll
    for (int j = 0; j < 5; ++j) {
        float d = 0.f;
        #pragma unroll
        for (int t = 0; t < 16; ++t) d += w[4][t] * w[j][t];
        #pragma unroll
        for (int msk = 16; msk >= 1; msk >>= 1) d += __shfl_xor(d, msk, 32);
        dots[j] = d * 0.044194173824159216f;   // 1/sqrt(512)
    }
    float mx = dots[0];
    #pragma unroll
    for (int j = 1; j < 5; ++j) mx = fmaxf(mx, dots[j]);
    float e[5], sum = 0.f;
    #pragma unroll
    for (int j = 0; j < 5; ++j) { e[j] = __expf(dots[j] - mx); sum += e[j]; }
    float inv = 1.f / sum;

    unsigned outw[8];
    #pragma unroll
    for (int t = 0; t < 8; ++t) {
        float v0 = 0.f, v1 = 0.f;
        #pragma unroll
        for (int j = 0; j < 5; ++j) {
            float wj = e[j] * inv;
            v0 += wj * w[j][2*t];
            v1 += wj * w[j][2*t+1];
        }
        outw[t] = pack2(v0, v1);
    }
    unsigned short* op = outq + ((size_t)n * 128 + s) * 512 + d0;
    *(uint4*)op       = make_uint4(outw[0], outw[1], outw[2], outw[3]);
    *(uint4*)(op + 8) = make_uint4(outw[4], outw[5], outw[6], outw[7]);
}

// ---------------------------------------------------------------------------
// MHA over S=128: one block per (batch n, head h). q,k,v,o: bf16 (256,128,512)
// Dynamic LDS: Q[128][72] K[128][72] Vt[64][136] P[128][136] bf16 + Sc[128][132] f32
// ---------------------------------------------------------------------------
__global__ __launch_bounds__(256) void mha1_kernel(
    const unsigned short* __restrict__ q, const unsigned short* __restrict__ k,
    const unsigned short* __restrict__ v, unsigned short* __restrict__ o)
{
    extern __shared__ __align__(16) unsigned char smem_raw[];
    unsigned short* Qs = (unsigned short*)smem_raw;      // 128*72
    unsigned short* Ks = Qs + 128 * 72;                  // 128*72
    unsigned short* Vt = Ks + 128 * 72;                  // 64*136 (feature-major)
    unsigned short* Pb = Vt + 64 * 136;                  // 128*136
    float*          Sc = (float*)(Pb + 128 * 136);       // 128*132

    const int tid  = threadIdx.x;
    const int lane = tid & 31;
    const int wave = tid >> 5;
    const int n = blockIdx.x >> 3;
    const int h = blockIdx.x & 7;
    const size_t hb = (size_t)h * 64;

    #pragma unroll
    for (int i = 0; i < 8; ++i) {
        int g   = tid + i * 256;
        int row = g >> 4;
        int c4  = (g & 15) * 4;
        size_t src = ((size_t)n * 128 + row) * 512 + hb + c4;
        *(uint2*)&Qs[row * 72 + c4] = *(const uint2*)(q + src);
        *(uint2*)&Ks[row * 72 + c4] = *(const uint2*)(k + src);
        uint2 vv = *(const uint2*)(v + src);
        Vt[(c4 + 0) * 136 + row] = (unsigned short)(vv.x & 0xffffu);
        Vt[(c4 + 1) * 136 + row] = (unsigned short)(vv.x >> 16);
        Vt[(c4 + 2) * 136 + row] = (unsigned short)(vv.y & 0xffffu);
        Vt[(c4 + 3) * 136 + row] = (unsigned short)(vv.y >> 16);
    }
    __syncthreads();

    const int fm = lane & 15;
    const int lh = lane >> 4;
    const int kA = lh * 8;
    const int kB = lh * 8;

    { // scores = Q*K^T / 8 ; M=N=128, K=64
        const int wr = wave >> 2, wc = wave & 3;
        const int mW = wr * 64, nW = wc * 32;
        v8f acc[4][2];
        #pragma unroll
        for (int i = 0; i < 4; ++i)
            #pragma unroll
            for (int j = 0; j < 2; ++j) acc[i][j] = v8f_zero();
        #pragma unroll
        for (int kb = 0; kb < 64; kb += 32) {
            Frag16 af[4], bf[2];
            #pragma unroll
            for (int i = 0; i < 4; ++i) {
                const unsigned short* p = &Qs[(mW + i * 16 + fm) * 72 + kb + kA];
                af[i].q[0] = *(const uint4*)p;
                af[i].q[1] = *(const uint4*)(p + 16);
            }
            #pragma unroll
            for (int j = 0; j < 2; ++j) {
                const unsigned short* p = &Ks[(nW + j * 16 + fm) * 72 + kb + kB];
                bf[j].q[0] = *(const uint4*)p;
                bf[j].q[1] = *(const uint4*)(p + 16);
            }
            #pragma unroll
            for (int i = 0; i < 4; ++i)
                #pragma unroll
                for (int j = 0; j < 2; ++j)
                    acc[i][j] = wmma_bf16(af[i], bf[j], acc[i][j]);
        }
        #pragma unroll
        for (int i = 0; i < 4; ++i)
            #pragma unroll
            for (int j = 0; j < 2; ++j)
                #pragma unroll
                for (int r = 0; r < 8; ++r)
                    Sc[(mW + i*16 + lh*8 + r) * 132 + nW + j*16 + fm] =
                        acc[i][j][r] * 0.125f;   // 1/sqrt(64)
    }
    __syncthreads();

    // softmax: wave handles rows wave*16..+15, 4 cols per lane
    for (int rr = 0; rr < 16; ++rr) {
        int row = wave * 16 + rr;
        float vals[4];
        #pragma unroll
        for (int t = 0; t < 4; ++t) vals[t] = Sc[row * 132 + lane + t * 32];
        float mx = fmaxf(fmaxf(vals[0], vals[1]), fmaxf(vals[2], vals[3]));
        #pragma unroll
        for (int msk = 16; msk >= 1; msk >>= 1) mx = fmaxf(mx, __shfl_xor(mx, msk, 32));
        float sum = 0.f;
        #pragma unroll
        for (int t = 0; t < 4; ++t) { vals[t] = __expf(vals[t] - mx); sum += vals[t]; }
        #pragma unroll
        for (int msk = 16; msk >= 1; msk >>= 1) sum += __shfl_xor(sum, msk, 32);
        float inv = 1.f / sum;
        #pragma unroll
        for (int t = 0; t < 4; ++t)
            Pb[row * 136 + lane + t * 32] = f2bf(vals[t] * inv);
    }
    __syncthreads();

    { // O = P*V ; M=128, N=64, K=128; wave tile 64x16
        const int wr = wave >> 2, wc = wave & 3;
        const int mW = wr * 64, nW = wc * 16;
        v8f acc[4];
        #pragma unroll
        for (int i = 0; i < 4; ++i) acc[i] = v8f_zero();
        #pragma unroll
        for (int kb = 0; kb < 128; kb += 32) {
            Frag16 af[4], bf1;
            #pragma unroll
            for (int i = 0; i < 4; ++i) {
                const unsigned short* p = &Pb[(mW + i * 16 + fm) * 136 + kb + kA];
                af[i].q[0] = *(const uint4*)p;
                af[i].q[1] = *(const uint4*)(p + 16);
            }
            const unsigned short* p = &Vt[(nW + fm) * 136 + kb + kB];
            bf1.q[0] = *(const uint4*)p;
            bf1.q[1] = *(const uint4*)(p + 16);
            #pragma unroll
            for (int i = 0; i < 4; ++i) acc[i] = wmma_bf16(af[i], bf1, acc[i]);
        }
        #pragma unroll
        for (int i = 0; i < 4; ++i)
            #pragma unroll
            for (int r = 0; r < 8; ++r)
                o[((size_t)n * 128 + mW + i*16 + lh*8 + r) * 512 + hb + nW + fm] =
                    f2bf(acc[i][r]);
    }
}

// ---------------------------------------------------------------------------
// MHA over A=16: one block per m = s*16+b (2048 blocks), one wave per head.
// Writes x2 already permuted into final (a,b,s,d) row order.
// ---------------------------------------------------------------------------
__global__ __launch_bounds__(256) void mha2_kernel(
    const unsigned short* __restrict__ x1, unsigned short* __restrict__ x2)
{
    extern __shared__ __align__(16) unsigned char smem_raw[];
    unsigned short* Xs   = (unsigned short*)smem_raw;  // [16][520]
    unsigned short* Vt2  = Xs + 16 * 520;              // [8][64][40] (k>=16 zero)
    unsigned short* Pbuf = Vt2 + 8 * 64 * 40;          // [8][16][40] (k>=16 zero)

    const int tid  = threadIdx.x;
    const int lane = tid & 31;
    const int h    = tid >> 5;          // wave == head
    const int m = blockIdx.x;
    const int s = m >> 4;
    const int b = m & 15;

    #pragma unroll
    for (int i = 0; i < 8; ++i) {
        int g   = tid + i * 256;
        int row = g >> 7;               // a = 0..15
        int c4  = (g & 127) * 4;        // 0..508
        size_t src = (((size_t)(row * 16 + b)) * 128 + s) * 512 + c4;
        *(uint2*)&Xs[row * 520 + c4] = *(const uint2*)(x1 + src);
    }
    #pragma unroll
    for (int i = 0; i < 20; ++i) Pbuf[tid + i * 256] = 0;   // 8*16*40 = 5120
    __syncthreads();

    unsigned short* vt  = Vt2 + h * 64 * 40;
    unsigned short* pbh = Pbuf + h * 16 * 40;
    #pragma unroll
    for (int t = 0; t < 64; ++t) {       // build V^T (64 feats x 32 k, k<16 valid)
        int e  = lane + t * 32;
        int d  = e >> 5;
        int kk = e & 31;
        vt[d * 40 + kk] = (kk < 16) ? Xs[kk * 520 + h * 64 + d] : (unsigned short)0;
    }

    const int fm = lane & 15;
    const int lh = lane >> 4;
    const int kA = lh * 8;
    const int kB = lh * 8;
    const int cb = h * 64;

    // scores 16x16, K=64
    v8f sc = v8f_zero();
    #pragma unroll
    for (int kb = 0; kb < 64; kb += 32) {
        Frag16 af, bf1;
        const unsigned short* pa = &Xs[fm * 520 + cb + kb + kA];
        af.q[0] = *(const uint4*)pa;
        af.q[1] = *(const uint4*)(pa + 16);
        const unsigned short* pb = &Xs[fm * 520 + cb + kb + kB];
        bf1.q[0] = *(const uint4*)pb;
        bf1.q[1] = *(const uint4*)(pb + 16);
        sc = wmma_bf16(af, bf1, sc);
    }
    // softmax: score row m lives in one 16-lane half (C layout), reduce via xor<16
    #pragma unroll
    for (int r = 0; r < 8; ++r) {
        float vv = sc[r] * 0.125f;
        float mx = vv;
        #pragma unroll
        for (int msk = 8; msk >= 1; msk >>= 1) mx = fmaxf(mx, __shfl_xor(mx, msk, 32));
        float e = __expf(vv - mx);
        float sum = e;
        #pragma unroll
        for (int msk = 8; msk >= 1; msk >>= 1) sum += __shfl_xor(sum, msk, 32);
        pbh[(r + lh * 8) * 40 + fm] = f2bf(e / sum);
    }
    __syncthreads();

    // O = P*V : 16x64, K=32 (zero padded beyond 16)
    Frag16 af;
    const unsigned short* pa = &pbh[fm * 40 + kA];
    af.q[0] = *(const uint4*)pa;
    af.q[1] = *(const uint4*)(pa + 16);
    #pragma unroll
    for (int jn = 0; jn < 4; ++jn) {
        Frag16 bf1;
        const unsigned short* pb = &vt[(jn * 16 + fm) * 40 + kB];
        bf1.q[0] = *(const uint4*)pb;
        bf1.q[1] = *(const uint4*)(pb + 16);
        v8f acc = v8f_zero();
        acc = wmma_bf16(af, bf1, acc);
        #pragma unroll
        for (int r = 0; r < 8; ++r) {
            int a = r + lh * 8;
            size_t off = (((size_t)(a * 16 + b)) * 128 + s) * 512 + cb + jn * 16 + fm;
            x2[off] = f2bf(acc[r]);
        }
    }
}

// ---------------------------------------------------------------------------
extern "C" void kernel_launch(void* const* d_in, const int* in_sizes, int n_in,
                              void* d_out, int out_size, void* d_ws, size_t ws_size,
                              hipStream_t stream)
{
    const float* query = (const float*)d_in[0];
    const float* key_t = (const float*)d_in[1];
    const float* value = (const float*)d_in[2];
    // d_in[3] = mask (unused by reference)
    const float* pad_q = (const float*)d_in[4];
    const float* pad_k = (const float*)d_in[5];
    const float* Wcq = (const float*)d_in[6];
    const float* bcq = (const float*)d_in[7];
    const float* Wck = (const float*)d_in[8];
    const float* bck = (const float*)d_in[9];
    const float* Wv  = (const float*)d_in[10];
    const float* bv  = (const float*)d_in[11];
    const float* Wo  = (const float*)d_in[12];
    const float* bo  = (const float*)d_in[13];

    unsigned short* ws = (unsigned short*)d_ws;
    const size_t NY = (size_t)256 * 132 * 512;   // padded projection rows
    const size_t NX = (size_t)256 * 128 * 512;
    unsigned short* yq = ws;            // bf16 (256,132,512)
    unsigned short* yk = yq + NY;       // bf16 (256,132,512)
    unsigned short* vb = yk + NY;       // bf16 (256,128,512)
    unsigned short* qb = vb + NX;       // bf16 (256,128,512)
    unsigned short* kb = qb + NX;       // bf16 (256,128,512)
    unsigned short* x1 = yq;            // alias: yq dead after local_attn
    unsigned short* x2 = yk;            // alias: yk dead after local_attn

    dim3 blk(256);

    // local-branch projections (rows = 4 pad rows + 128 seq rows per batch)
    gemm_wmma_kernel<false,false><<<dim3(264, 4), blk, 0, stream>>>(
        query, pad_q, 4, 132, Wcq, bcq, yq, 33792, 512, 512);
    gemm_wmma_kernel<false,false><<<dim3(264, 4), blk, 0, stream>>>(
        key_t, pad_k, 4, 132, Wck, bck, yk, 33792, 512, 512);
    // value projection
    gemm_wmma_kernel<false,false><<<dim3(256, 4), blk, 0, stream>>>(
        value, nullptr, 0, 0, Wv, bv, vb, 32768, 512, 512);

    // local windowed attention
    local_attn_kernel<<<4096, blk, 0, stream>>>(yq, qb);
    local_attn_kernel<<<4096, blk, 0, stream>>>(yk, kb);

    // MHA over sequence
    size_t smem1 = (size_t)(128*72 + 128*72 + 64*136 + 128*136) * 2
                 + (size_t)128 * 132 * 4;   // ~153 KB < 320 KB WGP LDS
    mha1_kernel<<<2048, blk, smem1, stream>>>(qb, kb, vb, x1);

    // MHA over asset axis (writes x2 in final (a,b,s) row order)
    size_t smem2 = (size_t)(16*520 + 8*64*40 + 8*16*40) * 2;   // ~66 KB
    mha2_kernel<<<2048, blk, smem2, stream>>>(x1, x2);

    // output projection -> fp32 d_out
    gemm_wmma_kernel<true,true><<<dim3(256, 4), blk, 0, stream>>>(
        x2, nullptr, 0, 0, Wo, bo, d_out, 32768, 512, 512);
}
